// SelfAttention_46995532153494
// MI455X (gfx1250) — matrix operand order repaired and checked by reading, not compiled, not run on hardware
//
#include <hip/hip_runtime.h>
#include <math.h>

// ---------------------------------------------------------------------------
// MI455X (gfx1250) self-attention pipeline, f16 WMMA, wave32.
// ---------------------------------------------------------------------------

typedef __attribute__((ext_vector_type(16))) _Float16     v16h;
typedef __attribute__((ext_vector_type(8)))  float        v8f;
typedef __attribute__((ext_vector_type(4)))  unsigned int u32x4;

union Frag {
    v16h  h;
    u32x4 u[2];
};

#define C_DIM  256
#define N_TOK  32768
#define HEADS  8
#define HD     32
#define T_SEG  256
#define B_SEG  128

#if __has_builtin(__builtin_amdgcn_global_load_async_to_lds_b128)
#define HAVE_ASYNC_LDS 1
// exact pointee type per hipcc diagnostic: int __attribute__((vector_size(16)))
typedef int v128i __attribute__((vector_size(16)));
#define AS1 __attribute__((address_space(1)))
#define AS3 __attribute__((address_space(3)))
#endif

// ---------------------------------------------------------------------------
// f32 -> f16 conversion (x and the four weight matrices)
// ---------------------------------------------------------------------------
__global__ void cvt_f32_f16_kernel(const float* __restrict__ in,
                                   _Float16* __restrict__ out, int n) {
    int i = blockIdx.x * blockDim.x + threadIdx.x;
    if (i < n) out[i] = (_Float16)in[i];
}

// ---------------------------------------------------------------------------
// O = X (32768 x 256) @ W^T (256 x 256) + bias, via v_wmma_f32_16x16x32_f16.
// 4-wave block computes a 64x64 tile; each wave owns 16 rows x 64 cols.
// The block's 64x256 W band is staged once into LDS (async-to-LDS when the
// toolchain exposes the builtin) and shared by all 4 waves.
// modes: 0 = f16 row-major (K proj)
//        1 = f16 row-major, scaled by 1/sqrt(hd) (Q proj)
//        2 = f16 into transposed V layout [b,h,d,s] (V proj), packed stores
//        3 = f32 row-major into d_out (final projection)
// ---------------------------------------------------------------------------
#define WTS 264   // LDS W-row stride in halves: 528B -> lane lr hits bank lr*4

__global__ __launch_bounds__(128) void gemm_nt_kernel(
    const _Float16* __restrict__ X, const _Float16* __restrict__ W,
    const float* __restrict__ bias, _Float16* __restrict__ outh,
    float* __restrict__ outf, int mode)
{
    __shared__ __attribute__((aligned(16))) _Float16 Wt[64][WTS];

    const int tid   = threadIdx.x;
    const int lane  = tid & 31;
    const int wave  = tid >> 5;
    const int bm    = blockIdx.x >> 2;   // 512 M-tiles of 64
    const int bn    = blockIdx.x & 3;    // 4 N-tiles of 64
    const int mrow0 = bm * 64 + wave * 16;
    const int ncol0 = bn * 64;
    const int lr    = lane & 15;
    const int grp   = lane >> 4;

    // ---- stage W band (64 rows x 256 halves) into LDS ---------------------
    for (int c = tid; c < 2048; c += 128) {       // 2048 x 16B chunks
        int row = c >> 5;
        int off = (c & 31) * 8;
        const _Float16* g = W + (size_t)(ncol0 + row) * C_DIM + off;
        _Float16*       l = &Wt[row][off];
#ifdef HAVE_ASYNC_LDS
        __builtin_amdgcn_global_load_async_to_lds_b128(
            (AS1 v128i*)g, (AS3 v128i*)l, 0, 0);
#else
        *(u32x4*)l = *(const u32x4*)g;
#endif
    }
#ifdef HAVE_ASYNC_LDS
#if __has_builtin(__builtin_amdgcn_s_wait_asynccnt)
    __builtin_amdgcn_s_wait_asynccnt(0);
#else
    asm volatile("s_wait_asynccnt 0x0" ::: "memory");
#endif
#endif
    __syncthreads();

    // ---- K-pipelined WMMA main loop ---------------------------------------
    v8f acc[4] = {};
    Frag a, an;
    {
        const _Float16* ap = X + (size_t)(mrow0 + lr) * C_DIM + grp * 8;
        a.u[0] = *(const u32x4*)ap;
        a.u[1] = *(const u32x4*)(ap + 16);
    }
#pragma unroll
    for (int kc = 0; kc < 8; ++kc) {
        if (kc < 7) {   // preload next A fragment (global)
            const _Float16* ap = X + (size_t)(mrow0 + lr) * C_DIM
                                   + (kc + 1) * 32 + grp * 8;
            an.u[0] = *(const u32x4*)ap;
            an.u[1] = *(const u32x4*)(ap + 16);
            __builtin_prefetch(ap + 32, 0, 3);
        }
        Frag bf[4];     // all 4 B fragments up front (LDS)
#pragma unroll
        for (int nt = 0; nt < 4; ++nt) {
            const _Float16* bp = &Wt[nt * 16 + lr][kc * 32 + grp * 8];
            bf[nt].u[0] = *(const u32x4*)bp;
            bf[nt].u[1] = *(const u32x4*)(bp + 16);
        }
#pragma unroll
        for (int nt = 0; nt < 4; ++nt) {
            acc[nt] = __builtin_amdgcn_wmma_f32_16x16x32_f16(
                false, a.h, false, bf[nt].h, (short)0, acc[nt], false, false);
        }
        a = an;
    }

    // ---- epilogue ---------------------------------------------------------
    if (mode == 2) {
        // V projection -> transposed [b,h,d,s]; 8 consecutive s per lane-tile
        int m0   = mrow0 + grp * 8;
        int bseg = m0 >> 8, s0 = m0 & 255;
#pragma unroll
        for (int nt = 0; nt < 4; ++nt) {
            int col = ncol0 + nt * 16 + lr;
            int hh = col >> 5, d = col & 31;
            union { _Float16 h[8]; u32x4 u; } pk;
#pragma unroll
            for (int j = 0; j < 8; ++j)
                pk.h[j] = (_Float16)(acc[nt][j] + bias[col]);
            *(u32x4*)(outh + (size_t)((bseg * HEADS + hh) * HD + d) * T_SEG + s0) = pk.u;
        }
    } else {
#pragma unroll
        for (int nt = 0; nt < 4; ++nt) {
#pragma unroll
            for (int j = 0; j < 8; ++j) {
                int   m   = mrow0 + grp * 8 + j;
                int   col = ncol0 + nt * 16 + lr;
                float v   = acc[nt][j] + bias[col];
                if (mode == 0)
                    outh[(size_t)m * C_DIM + col] = (_Float16)v;
                else if (mode == 1)
                    outh[(size_t)m * C_DIM + col] = (_Float16)(v * 0.17677669529663687f);
                else
                    outf[(size_t)m * C_DIM + col] = v;
            }
        }
    }
}

// ---------------------------------------------------------------------------
// Fused attention: one block (2 waves) per (segment b, 16-row tile).
// Loops over the 8 heads: scores via WMMA -> LDS, softmax in LDS,
// head-mean accumulated in LDS f32 (deterministic), P@V via WMMA with the
// A-fragment read from LDS. Per-head attention never touches HBM.
// ---------------------------------------------------------------------------
#define SBS 258   // f32 score row stride (bank-conflict pad)
#define PBS 264   // f16 P row stride (16B-aligned rows)
#define MBS 258

__global__ __launch_bounds__(64) void attn_kernel(
    const _Float16* __restrict__ qh, const _Float16* __restrict__ kh,
    const _Float16* __restrict__ vt, _Float16* __restrict__ yh,
    float* __restrict__ attn_mean)
{
    __shared__ __attribute__((aligned(16))) float    Sbuf[16][SBS];
    __shared__ __attribute__((aligned(16))) _Float16 Pbuf[16][PBS];
    __shared__ __attribute__((aligned(16))) float    Mean[16][MBS];
    __shared__ float rmax[16][4];
    __shared__ float rsum[16][4];

    const int tid   = threadIdx.x;
    const int lane  = tid & 31;
    const int wave  = tid >> 5;
    const int lr    = lane & 15;
    const int grp   = lane >> 4;
    const int b     = blockIdx.x >> 4;   // segment
    const int t16   = blockIdx.x & 15;   // 16-row tile within segment
    const int trow0 = t16 * 16;

    for (int i = tid; i < 16 * 256; i += 64) Mean[i >> 8][i & 255] = 0.0f;
    __syncthreads();

    for (int h = 0; h < HEADS; ++h) {
        // ---- S = Q K^T (scale folded into Q) : pipelined over 8 s-tiles ---
        Frag qa;
        const _Float16* qp = qh + (size_t)(b * T_SEG + trow0 + lr) * C_DIM
                                + h * HD + grp * 8;
        qa.u[0] = *(const u32x4*)qp;
        qa.u[1] = *(const u32x4*)(qp + 16);

        auto loadK = [&](int st, Frag& f) {
            const _Float16* kp = kh + (size_t)(b * T_SEG + st * 16 + lr) * C_DIM
                                    + h * HD + grp * 8;
            f.u[0] = *(const u32x4*)kp;
            f.u[1] = *(const u32x4*)(kp + 16);
        };
        Frag kb[2];
        loadK(wave * 8, kb[0]);
#pragma unroll
        for (int i = 0; i < 8; ++i) {
            int st = wave * 8 + i;       // wave-uniform: EXEC stays all-ones
            if (i < 7) loadK(st + 1, kb[(i + 1) & 1]);
            v8f s = {};
            s = __builtin_amdgcn_wmma_f32_16x16x32_f16(
                false, qa.h, false, kb[i & 1].h, (short)0, s, false, false);
#pragma unroll
            for (int j = 0; j < 8; ++j)
                Sbuf[grp * 8 + j][st * 16 + lr] = s[j];
        }
        __syncthreads();

        // ---- softmax over 256 cols: 4 threads per row, 64 cols each -------
        const int r  = tid & 15;
        const int qq = tid >> 4;
        const int c0 = qq * 64;
        float mx = -3.4e38f;
        for (int i = 0; i < 64; ++i) mx = fmaxf(mx, Sbuf[r][c0 + i]);
        rmax[r][qq] = mx;
        __syncthreads();
        float rm = fmaxf(fmaxf(rmax[r][0], rmax[r][1]),
                         fmaxf(rmax[r][2], rmax[r][3]));
        float sm = 0.0f;
        for (int i = 0; i < 64; ++i) {
            float e = __expf(Sbuf[r][c0 + i] - rm);
            Sbuf[r][c0 + i] = e;
            sm += e;
        }
        rsum[r][qq] = sm;
        __syncthreads();
        float inv = 1.0f / (rsum[r][0] + rsum[r][1] + rsum[r][2] + rsum[r][3]);
        for (int i = 0; i < 64; ++i) {
            float p = Sbuf[r][c0 + i] * inv;
            Pbuf[r][c0 + i] = (_Float16)p;
            Mean[r][c0 + i] += 0.125f * p;       // head-mean, fixed order
        }
        __syncthreads();

        // ---- Y_h = P @ V : pipelined over 8 K-chunks ----------------------
        auto loadPV = [&](int kc, Frag& pa, Frag& vb) {
            const _Float16* pp = &Pbuf[lr][kc * 32 + grp * 8];
            pa.u[0] = *(const u32x4*)pp;
            pa.u[1] = *(const u32x4*)(pp + 16);
            const _Float16* vp = vt + (size_t)((b * HEADS + h) * HD + wave * 16 + lr) * T_SEG
                                    + kc * 32 + grp * 8;
            vb.u[0] = *(const u32x4*)vp;
            vb.u[1] = *(const u32x4*)(vp + 16);
        };
        Frag pa[2], vb[2];
        loadPV(0, pa[0], vb[0]);
        v8f acc = {};
#pragma unroll
        for (int kc = 0; kc < 8; ++kc) {
            if (kc < 7) loadPV(kc + 1, pa[(kc + 1) & 1], vb[(kc + 1) & 1]);
            acc = __builtin_amdgcn_wmma_f32_16x16x32_f16(
                false, pa[kc & 1].h, false, vb[kc & 1].h, (short)0, acc, false, false);
        }
#pragma unroll
        for (int j = 0; j < 8; ++j) {
            int t   = trow0 + grp * 8 + j;
            int col = h * HD + wave * 16 + lr;
            yh[(size_t)(b * T_SEG + t) * C_DIM + col] = (_Float16)acc[j];
        }
        __syncthreads();
    }

    // ---- emit head-mean attention block -----------------------------------
    for (int i = tid; i < 16 * 256; i += 64) {
        int r = i >> 8, c = i & 255;
        attn_mean[((size_t)b * T_SEG + trow0 + r) * T_SEG + c] = Mean[r][c];
    }
}

// ---------------------------------------------------------------------------
extern "C" void kernel_launch(void* const* d_in, const int* in_sizes, int n_in,
                              void* d_out, int out_size, void* d_ws, size_t ws_size,
                              hipStream_t stream) {
    (void)in_sizes; (void)n_in; (void)out_size; (void)ws_size;

    const float* x  = (const float*)d_in[0];
    // d_in[1] = batch (unused: equal-length segments, B=128 fixed)
    const float* Wq = (const float*)d_in[2];
    const float* bq = (const float*)d_in[3];
    const float* Wk = (const float*)d_in[4];
    const float* bk = (const float*)d_in[5];
    const float* Wv = (const float*)d_in[6];
    const float* bv = (const float*)d_in[7];
    const float* Wp = (const float*)d_in[8];
    const float* bp = (const float*)d_in[9];

    float* y_out    = (float*)d_out;
    float* attn_out = y_out + (size_t)N_TOK * C_DIM;

    // workspace layout (f16 buffers), ~80.5 MB total
    char* ws = (char*)d_ws;
    const size_t XB = (size_t)N_TOK * C_DIM * sizeof(_Float16);  // 16 MiB
    _Float16* x_h  = (_Float16*)(ws);
    _Float16* q_h  = (_Float16*)(ws + 1 * XB);
    _Float16* k_h  = (_Float16*)(ws + 2 * XB);
    _Float16* vt_h = (_Float16*)(ws + 3 * XB);   // [b,h,d,s]
    _Float16* y_h  = (_Float16*)(ws + 4 * XB);
    _Float16* wq_h = (_Float16*)(ws + 5 * XB);
    _Float16* wk_h = wq_h + C_DIM * C_DIM;
    _Float16* wv_h = wk_h + C_DIM * C_DIM;
    _Float16* wp_h = wv_h + C_DIM * C_DIM;

    const int NX = N_TOK * C_DIM;      // 8,388,608
    const int NW = C_DIM * C_DIM;      // 65,536
    cvt_f32_f16_kernel<<<(NX + 255) / 256, 256, 0, stream>>>(x,  x_h,  NX);
    cvt_f32_f16_kernel<<<(NW + 255) / 256, 256, 0, stream>>>(Wq, wq_h, NW);
    cvt_f32_f16_kernel<<<(NW + 255) / 256, 256, 0, stream>>>(Wk, wk_h, NW);
    cvt_f32_f16_kernel<<<(NW + 255) / 256, 256, 0, stream>>>(Wv, wv_h, NW);
    cvt_f32_f16_kernel<<<(NW + 255) / 256, 256, 0, stream>>>(Wp, wp_h, NW);

    // QKV projections (Q pre-scaled by 1/sqrt(hd); V stored transposed)
    gemm_nt_kernel<<<2048, 128, 0, stream>>>(x_h, wq_h, bq, q_h,  nullptr, 1);
    gemm_nt_kernel<<<2048, 128, 0, stream>>>(x_h, wk_h, bk, k_h,  nullptr, 0);
    gemm_nt_kernel<<<2048, 128, 0, stream>>>(x_h, wv_h, bv, vt_h, nullptr, 2);

    // fused scores + softmax + head-mean + P@V
    attn_kernel<<<B_SEG * 16, 64, 0, stream>>>(q_h, k_h, vt_h, y_h, attn_out);

    // output projection -> f32 d_out
    gemm_nt_kernel<<<2048, 128, 0, stream>>>(y_h, wp_h, bp, nullptr, y_out, 3);
}
